// DeepNCM_66872640799201
// MI455X (gfx1250) — matrix operand order repaired and checked by reading, not compiled, not run on hardware
//
#include <hip/hip_runtime.h>
#include <hip/hip_bf16.h>

#define N_ROWS  65536
#define FDIM    256
#define NCLS    1000
#define NCLSP   1008        // padded to 63*16 so B-tile loads need no clamping
#define NTILES  63
#define ALPHA_F 0.9f
#define SCALE_C 9.7676261042f   // sqrt(2) * ln(999)
#define EPS_N   1e-12f

#define LDS_ROW 264         // 256 bf16 + 8 pad -> 528B row stride, conflict-free frags

typedef __attribute__((ext_vector_type(16))) __bf16 v16bf;
typedef __attribute__((ext_vector_type(8)))  float  v8f;

union BF16Tile {
    v16bf        v;
    uint4        q[2];
    unsigned int u[8];
};

// round-to-nearest-even f32 -> bf16, packed pair into one dword
__device__ __forceinline__ unsigned int pack_bf16x2(float lo, float hi) {
    unsigned int ulo = __float_as_uint(lo);
    unsigned int uhi = __float_as_uint(hi);
    ulo += 0x7FFFu + ((ulo >> 16) & 1u);
    uhi += 0x7FFFu + ((uhi >> 16) & 1u);
    return (ulo >> 16) | (uhi & 0xFFFF0000u);
}

// Workgroup barrier that only drains this wave's LDS ops (no global-store
// fence): ds ordering + CDNA5 split barrier. __syncthreads() would insert
// s_wait_storecnt_dscnt 0 and serialize the output stores into the loop.
__device__ __forceinline__ void lds_barrier() {
    asm volatile("s_wait_dscnt 0x0\n\t"
                 "s_barrier_signal -1\n\t"
                 "s_barrier_wait -1" ::: "memory");
}

// ---------------------------------------------------------------------------
// Kernel 1: zero sums/counts, convert means f32 -> bf16 (padded to NCLSP rows)
// ---------------------------------------------------------------------------
__global__ __launch_bounds__(256) void k_init(const float* __restrict__ means,
                                              unsigned short* __restrict__ meansBf,
                                              float* __restrict__ sums,
                                              float* __restrict__ counts) {
    int i = blockIdx.x * 256 + threadIdx.x;      // [0, NCLSP*FDIM)
    int c = i >> 8;                              // / FDIM
    int f = i & 255;
    float m = (c < NCLS) ? means[c * FDIM + f] : 0.0f;
    unsigned int u = __float_as_uint(m);
    u += 0x7FFFu + ((u >> 16) & 1u);
    meansBf[i] = (unsigned short)(u >> 16);
    if (i < NCLS * FDIM) sums[i] = 0.0f;
    if (i < NCLS)        counts[i] = 0.0f;
}

// ---------------------------------------------------------------------------
// Kernel 2: one wave32 per row: inv-norm + atomic segment sums/counts
// ---------------------------------------------------------------------------
__global__ __launch_bounds__(256) void k_rownorm(const float* __restrict__ x,
                                                 const int* __restrict__ labels,
                                                 float* __restrict__ inv_norm,
                                                 float* __restrict__ sums,
                                                 float* __restrict__ counts) {
    int wave = threadIdx.x >> 5;
    int lane = threadIdx.x & 31;
    int row  = blockIdx.x * 8 + wave;

    const float* xr = x + (size_t)row * FDIM + lane * 8;
    float4 a = *(const float4*)xr;
    float4 b = *(const float4*)(xr + 4);
    float ss = a.x*a.x + a.y*a.y + a.z*a.z + a.w*a.w
             + b.x*b.x + b.y*b.y + b.z*b.z + b.w*b.w;
#pragma unroll
    for (int off = 16; off > 0; off >>= 1)
        ss += __shfl_xor(ss, off, 32);

    float inv = 1.0f / fmaxf(sqrtf(ss), EPS_N);
    int   lbl = labels[row];
    float* s = sums + (size_t)lbl * FDIM + lane * 8;
    atomicAdd(s + 0, a.x * inv);
    atomicAdd(s + 1, a.y * inv);
    atomicAdd(s + 2, a.z * inv);
    atomicAdd(s + 3, a.w * inv);
    atomicAdd(s + 4, b.x * inv);
    atomicAdd(s + 5, b.y * inv);
    atomicAdd(s + 6, b.z * inv);
    atomicAdd(s + 7, b.w * inv);
    if (lane == 0) {
        inv_norm[row] = inv;
        atomicAdd(counts + lbl, 1.0f);
    }
}

// ---------------------------------------------------------------------------
// Kernel 3: cos_sims = (x * inv_norm) @ meansBf^T * SCALE via v_wmma bf16.
// One wave per 16-row M-tile (A in 64 VGPRs). 8 waves share each 16-class
// B tile via a double-buffered LDS stage; all 8 B fragments are pulled from
// LDS into registers before the WMMA chain so ds_load latency is pipelined.
// ---------------------------------------------------------------------------
__global__ __launch_bounds__(256) void k_gemm(const float* __restrict__ x,
                                              const float* __restrict__ inv_norm,
                                              const unsigned short* __restrict__ meansBf,
                                              float* __restrict__ cos_out) {
    __shared__ __align__(16) unsigned short lds[2][16 * LDS_ROW];

    int tid   = threadIdx.x;
    int wave  = tid >> 5;
    int lane  = tid & 31;
    int mtile = blockIdx.x * 8 + wave;
    int m0    = mtile * 16;
    int mrow  = m0 + (lane & 15);

    // ISA A layout (16-bit, 16x32): lanes 0-15 hold K [k0..k0+7, k0+16..k0+23],
    // lanes 16-31 hold K [k0+8..k0+15, k0+24..k0+31].
    int koffA = (lane < 16) ? 0 : 8;
    // B layout matches A: lanes 0-15 hold K [k0..k0+7, k0+16..k0+23], lanes 16-31 K [k0+8..k0+15, k0+24..k0+31].
    int koffB = (lane < 16) ? 0 : 8;      // in bf16 elements

    float inv = inv_norm[mrow];
    const float* xr = x + (size_t)mrow * FDIM;

    BF16Tile A[8];
#pragma unroll
    for (int kc = 0; kc < 8; ++kc) {
        int k0 = kc * 32 + koffA;
        float4 f0 = *(const float4*)(xr + k0);
        float4 f1 = *(const float4*)(xr + k0 + 4);
        float4 f2 = *(const float4*)(xr + k0 + 16);
        float4 f3 = *(const float4*)(xr + k0 + 20);
        A[kc].u[0] = pack_bf16x2(f0.x * inv, f0.y * inv);
        A[kc].u[1] = pack_bf16x2(f0.z * inv, f0.w * inv);
        A[kc].u[2] = pack_bf16x2(f1.x * inv, f1.y * inv);
        A[kc].u[3] = pack_bf16x2(f1.z * inv, f1.w * inv);
        A[kc].u[4] = pack_bf16x2(f2.x * inv, f2.y * inv);
        A[kc].u[5] = pack_bf16x2(f2.z * inv, f2.w * inv);
        A[kc].u[6] = pack_bf16x2(f3.x * inv, f3.y * inv);
        A[kc].u[7] = pack_bf16x2(f3.z * inv, f3.w * inv);
    }

    // Cooperative B staging: thread t owns a 32B chunk of the 16x256 bf16 tile.
    int cls_t  = tid >> 4;                 // class-in-tile 0..15
    int chnk_t = tid & 15;                 // 32B chunk-in-row 0..15
    const unsigned short* gbase = meansBf + (size_t)cls_t * FDIM + chnk_t * 16;
    unsigned short* lbase = &lds[0][0];    // indexed by (nt&1)*16*LDS_ROW
    int stoff = cls_t * LDS_ROW + chnk_t * 16;   // ushort units

    int ncol = lane & 15;

    // Preload tile 0 into registers.
    uint4 r0 = *(const uint4*)(gbase);
    uint4 r1 = *(const uint4*)(gbase + 8);

    for (int nt = 0; nt < NTILES; ++nt) {
        unsigned short* buf = lbase + (nt & 1) * (16 * LDS_ROW);
        *(uint4*)(buf + stoff)     = r0;
        *(uint4*)(buf + stoff + 8) = r1;
        lds_barrier();                     // buf[nt&1] ready for all waves

        if (nt + 1 < NTILES) {             // overlap next tile's global loads
            const unsigned short* g = gbase + (size_t)(nt + 1) * 16 * FDIM;
            r0 = *(const uint4*)(g);
            r1 = *(const uint4*)(g + 8);
        }

        // Pull all 8 B fragments first: 16 ds_load_b128 issue back-to-back and
        // the WMMA chain drains them with staged (non-zero) dscnt waits.
        const unsigned short* br = buf + ncol * LDS_ROW + koffB;
        BF16Tile B[8];
#pragma unroll
        for (int kc = 0; kc < 8; ++kc) {
            B[kc].q[0] = *(const uint4*)(br + kc * 32);
            B[kc].q[1] = *(const uint4*)(br + kc * 32 + 16);
        }

        v8f acc = {};
#pragma unroll
        for (int kc = 0; kc < 8; ++kc) {
            acc = __builtin_amdgcn_wmma_f32_16x16x32_bf16(
                      /*neg_a=*/false, A[kc].v, /*neg_b=*/false, B[kc].v,
                      /*c_mod=*/(short)0, acc, /*reuse_a=*/false, /*reuse_b=*/false);
        }

        int n = nt * 16 + ncol;
        if (n < NCLS) {
            // C/D layout: lane<16 -> VGPR r = D[M=r][N=lane]; lane>=16 -> D[M=8+r][N=lane-16]
            int mbase = m0 + ((lane < 16) ? 0 : 8);
            float* o = cos_out + (size_t)mbase * NCLS + n;
#pragma unroll
            for (int r = 0; r < 8; ++r)
                o[(size_t)r * NCLS] = acc[r] * SCALE_C;
        }
        // One barrier per iteration suffices: a wave reaching iteration nt+1's
        // ds_store has passed iteration nt's barrier, which implies every wave
        // finished its LDS reads of buf[(nt+1)&1] back at iteration nt-1.
    }
}

// ---------------------------------------------------------------------------
// Kernel 4: EMA update of means
// ---------------------------------------------------------------------------
__global__ __launch_bounds__(256) void k_ema(const float* __restrict__ means,
                                             const float* __restrict__ sums,
                                             const float* __restrict__ counts,
                                             float* __restrict__ out_means) {
    int i = blockIdx.x * 256 + threadIdx.x;   // < NCLS*FDIM
    int c = i >> 8;
    float cnt = counts[c];
    float m   = means[i];
    float running = (cnt > 1e-7f) ? (sums[i] / fmaxf(cnt, 1e-7f)) : m;
    out_means[i] = ALPHA_F * m + (1.0f - ALPHA_F) * running;
}

extern "C" void kernel_launch(void* const* d_in, const int* in_sizes, int n_in,
                              void* d_out, int out_size, void* d_ws, size_t ws_size,
                              hipStream_t stream) {
    const float* x      = (const float*)d_in[0];
    const int*   labels = (const int*)d_in[1];
    const float* means  = (const float*)d_in[2];

    float* cos_out   = (float*)d_out;
    float* new_means = (float*)d_out + (size_t)N_ROWS * NCLS;

    char* ws = (char*)d_ws;
    float*          inv_norm = (float*)ws;                       // 65536*4   = 256 KB
    float*          counts   = (float*)(ws + 262144);            // 1000*4 (4 KB slot)
    float*          sums     = (float*)(ws + 266240);            // 1000*256*4 = 1 MB
    unsigned short* meansBf  = (unsigned short*)(ws + 1290240);  // 1008*256*2 = 504 KB

    k_init   <<<(NCLSP * FDIM) / 256, 256, 0, stream>>>(means, meansBf, sums, counts);
    k_rownorm<<<N_ROWS / 8,            256, 0, stream>>>(x, labels, inv_norm, sums, counts);
    k_gemm   <<<N_ROWS / 16 / 8,       256, 0, stream>>>(x, inv_norm, meansBf, cos_out);
    k_ema    <<<(NCLS * FDIM) / 256,   256, 0, stream>>>(means, sums, counts, new_means);
}
